// ScaledDotProductAttention_86380382257276
// MI455X (gfx1250) — matrix operand order repaired and checked by reading, not compiled, not run on hardware
//
#include <hip/hip_runtime.h>

// Shapes fixed by the reference: B=4, H=16, S=2048, D=64
#define BATCH   4
#define NHEADS  16
#define SLEN    2048
#define DHEAD   64
#define NEGINF  (-1.0e14f)

typedef __attribute__((ext_vector_type(16))) __bf16 v16bf;
typedef __attribute__((ext_vector_type(8)))  float  v8f;
typedef __attribute__((ext_vector_type(4)))  float  nv4f;

__device__ __forceinline__ v8f wmma_bf16(v16bf a, v16bf b, v8f c) {
  // (neg_a, A, neg_b, B, c_mod, C, reuse_a, reuse_b)
  return __builtin_amdgcn_wmma_f32_16x16x32_bf16(false, a, false, b, (short)0, c,
                                                 false, false);
}

// A-layout element -> K index (16-bit A 16x32):
//   VGPR 0-3: lanes0-15 K=0..7,  lanes16-31 K=8..15
//   VGPR 4-7: lanes0-15 K=16..23, lanes16-31 K=24..31
__device__ __forceinline__ int a_kidx(int e, int lh) {
  return lh * 8 + e + ((e < 8) ? 0 : 8);
}

// Non-temporal 2x float4 store of 8 consecutive floats (streamed, bypass L2 reuse)
__device__ __forceinline__ void nt_store8(float* p, const float* v) {
  nv4f a = {v[0], v[1], v[2], v[3]};
  nv4f b = {v[4], v[5], v[6], v[7]};
  __builtin_nontemporal_store(a, (nv4f*)p);
  __builtin_nontemporal_store(b, (nv4f*)(p + 4));
}

__global__ __launch_bounds__(32)
void attn_fused_kernel(const float* __restrict__ Q, const float* __restrict__ K,
                       const float* __restrict__ V, const int* __restrict__ MK,
                       float* __restrict__ out, float* __restrict__ attn) {
  const int lane = threadIdx.x & 31;
  const int lh   = lane >> 4;     // lane half (0/1)
  const int l16  = lane & 15;

  const int nqt = SLEN / 16;                 // 128 query tiles per (b,h)
  const int bid = blockIdx.x;
  const int bh  = bid / nqt;
  const int qt  = bid - bh * nqt;
  const int b   = bh / NHEADS;
  const int q0  = qt * 16;

  const size_t bh_off = (size_t)bh * SLEN * DHEAD;
  const float* Kb = K + bh_off;
  const float* Vb = V + bh_off;
  const int*   Mrow = MK   + ((size_t)b  * SLEN + (q0 + l16)) * SLEN;
  float*       Arow = attn + ((size_t)bh * SLEN + (q0 + l16)) * SLEN;
  float*       Orow = out  + ((size_t)bh * SLEN + (q0 + l16)) * DHEAD;

  // ---- Load Q tile once in B-layout (column q = l16; element e -> d = dbase + a_kidx(e, lh))
  // Fold the 1/sqrt(64) score scale into Q here (once) instead of per score tile.
  const float* qrow = Q + bh_off + (size_t)(q0 + l16) * DHEAD;
  v16bf bq0, bq1;
#pragma unroll
  for (int e = 0; e < 16; ++e) {
    bq0[e] = (__bf16)(qrow[a_kidx(e, lh)] * 0.125f);
    bq1[e] = (__bf16)(qrow[32 + a_kidx(e, lh)] * 0.125f);
  }

  // A-layout loader for a 16-row x 32-col fp32 block (rows strided DHEAD floats)
  auto loadA = [&](const float* row0) -> v16bf {
    v16bf a;
#pragma unroll
    for (int i = 0; i < 8; ++i) {
      const int kb = lh * 8 + 2 * i + ((i < 4) ? 0 : 8);
      float2 f = *(const float2*)(row0 + kb);
      a[2 * i]     = (__bf16)f.x;
      a[2 * i + 1] = (__bf16)f.y;
    }
    return a;
  };

  // One 16x16 S^T tile (keys on VGPR axis, queries on lanes), masked.
  // s[i] = score(key = j*16 + lh*8 + i, q = l16); scale already folded into Q.
  auto score_tile = [&](int j, float* s) {
    const float* krow = Kb + (size_t)(j * 16 + l16) * DHEAD;
    v16bf a0 = loadA(krow);
    v16bf a1 = loadA(krow + 32);
    v8f acc = {};
    acc = wmma_bf16(a0, bq0, acc);
    acc = wmma_bf16(a1, bq1, acc);
    const int* mp = Mrow + j * 16 + lh * 8;
    int4 ma = *(const int4*)(mp);
    int4 mb = *(const int4*)(mp + 4);
    const int mi[8] = {ma.x, ma.y, ma.z, ma.w, mb.x, mb.y, mb.z, mb.w};
#pragma unroll
    for (int i = 0; i < 8; ++i)
      s[i] = (mi[i] == 0) ? NEGINF : acc[i];
  };

  // ================= Pass 1: exact row max m and row sum l (online) =========
  float m = -3.0e38f, lsum = 0.0f;
  for (int j = 0; j < nqt; ++j) {
    float s[8];
    score_tile(j, s);
    float tm = s[0];
#pragma unroll
    for (int i = 1; i < 8; ++i) tm = fmaxf(tm, s[i]);
    tm = fmaxf(tm, __shfl_xor(tm, 16, 32));
    const float mn = fmaxf(m, tm);
    float ls = 0.0f;
#pragma unroll
    for (int i = 0; i < 8; ++i) ls += __expf(s[i] - mn);
    ls += __shfl_xor(ls, 16, 32);
    lsum = lsum * __expf(m - mn) + ls;
    m = mn;
  }
  // Fold softmax normalization into the exponent: exp(s - m)/l == exp(s - (m + ln l))
  const float mshift = m + __logf(lsum);

  // ================= Pass 2: write attn (NT), accumulate O^T = V^T * P^T ====
  v8f o0 = {}, o1 = {}, o2 = {}, o3 = {};
  for (int t = 0; t < SLEN / 32; ++t) {
    float p0[8], p1[8];
#pragma unroll
    for (int sub = 0; sub < 2; ++sub) {
      const int j = 2 * t + sub;
      float s[8];
      score_tile(j, s);
      float* p = sub ? p1 : p0;
#pragma unroll
      for (int i = 0; i < 8; ++i) p[i] = __expf(s[i] - mshift);
      // attn is written once, never re-read: stream it past L2 (NT) so the
      // K/V/mask working sets stay resident for the cross-tile reuse.
      nt_store8(Arow + j * 16 + lh * 8, p);
    }
    // Build B-layout P^T for 32 keys: lanes0-15 need keys 0..15 of query l16,
    // lanes16-31 need keys 16..31. Half-swap via shfl_xor(16).
    float sw0[8], sw1[8];
#pragma unroll
    for (int i = 0; i < 8; ++i) {
      sw0[i] = __shfl_xor(p0[i], 16, 32);
      sw1[i] = __shfl_xor(p1[i], 16, 32);
    }
    v16bf bp;
#pragma unroll
    for (int e = 0; e < 16; ++e) {
      const float lo = (e < 8) ? p0[e]  : sw0[e - 8];
      const float hi = (e < 8) ? sw1[e] : p1[e - 8];
      bp[e] = (__bf16)(lh ? hi : lo);
    }
    // V^T chunks in A-layout: M = d (= c*16 + l16), K = key (lh*16 + e mapping)
    const float* vbase = Vb + (size_t)(t * 32) * DHEAD + l16;
#pragma unroll
    for (int c = 0; c < 4; ++c) {
      const float* vcol = vbase + c * 16;
      v16bf av;
#pragma unroll
      for (int e = 0; e < 16; ++e)
        av[e] = (__bf16)vcol[(size_t)(lh * 16 + e) * DHEAD];
      v8f* oc = (c == 0) ? &o0 : (c == 1) ? &o1 : (c == 2) ? &o2 : &o3;
      *oc = wmma_bf16(av, bp, *oc);
    }
  }

  // Store O (NT): o_c element i = O[q = l16][d = c*16 + lh*8 + i]
#pragma unroll
  for (int c = 0; c < 4; ++c) {
    const v8f oc = (c == 0) ? o0 : (c == 1) ? o1 : (c == 2) ? o2 : o3;
    float tmp[8] = {oc[0], oc[1], oc[2], oc[3], oc[4], oc[5], oc[6], oc[7]};
    nt_store8(Orow + c * 16 + lh * 8, tmp);
  }
}

extern "C" void kernel_launch(void* const* d_in, const int* in_sizes, int n_in,
                              void* d_out, int out_size, void* d_ws, size_t ws_size,
                              hipStream_t stream) {
  const float* Q  = (const float*)d_in[0];
  const float* K  = (const float*)d_in[1];
  const float* V  = (const float*)d_in[2];
  const int*   MK = (const int*)d_in[3];

  float* out  = (float*)d_out;
  float* attn = out + (size_t)BATCH * NHEADS * SLEN * DHEAD;  // tuple: (out, attn)

  const int blocks = BATCH * NHEADS * (SLEN / 16);  // 8192 tiles, 1 wave each
  attn_fused_kernel<<<blocks, 32, 0, stream>>>(Q, K, V, MK, out, attn);
}
